// NNUE_72971494359493
// MI455X (gfx1250) — matrix operand order repaired and checked by reading, not compiled, not run on hardware
//
#include <hip/hip_runtime.h>

// ---------------------------------------------------------------------------
// NNUE forward pass for MI455X (gfx1250, wave32, WMMA + async-to-LDS).
//
// Stage 0: pre-convert ft_w f32 -> f16 into workspace (async path only).
// Stage 1: feature transformer  C[4096][256] = feat[4096][40960] @ ft_w^T
//          (rows 0..2047 = white, 2048..4095 = black), f16 WMMA, f32 accum,
//          split-K=8 with hardware f32 atomics into workspace.
//          Double-buffered LDS B-tile; staging via GLOBAL_LOAD_ASYNC_TO_LDS
//          (ASYNCcnt) when workspace allows, VALU-convert path otherwise.
// Stage 2: per-position tail (stm mix, clamps, 32x512 / 32x32 / 1x32 layers).
// ---------------------------------------------------------------------------

#define F_DIM     40960
#define BATCH     2048
#define M_DIM     256
#define KSPLIT    8
#define KCHUNK    (F_DIM / KSPLIT)   // 5120
#define KSTEP     32
#define NSTEPS    (KCHUNK / KSTEP)   // 160
#define ROWBLOCKS 32                 // 4096 rows / 128 per workgroup
#define BS_STRIDE 40                 // padded f16 stride per ft_w row in LDS

typedef __attribute__((ext_vector_type(16))) _Float16 v16h;
typedef __attribute__((ext_vector_type(8)))  _Float16 v8h;
typedef __attribute__((ext_vector_type(8)))  float    v8f;
typedef __attribute__((ext_vector_type(4)))  float    fvec4;

__device__ __forceinline__ float clamp01(float x) {
    return fminf(fmaxf(x, 0.0f), 1.0f);
}

__global__ __launch_bounds__(256) void nnue_zero_kernel(float* __restrict__ p, int n) {
    int i = blockIdx.x * blockDim.x + threadIdx.x;
    if (i < n) p[i] = 0.0f;
}

// ft_w f32 -> f16 (8 elements per thread).
__global__ __launch_bounds__(256) void nnue_cvt_kernel(
    const float* __restrict__ src, _Float16* __restrict__ dst, int n8)
{
    int i = blockIdx.x * blockDim.x + threadIdx.x;
    if (i < n8) {
        const fvec4* s4 = (const fvec4*)(src + (size_t)i * 8);
        fvec4 v0 = s4[0], v1 = s4[1];
        v8h h;
        h[0] = (_Float16)v0[0]; h[1] = (_Float16)v0[1];
        h[2] = (_Float16)v0[2]; h[3] = (_Float16)v0[3];
        h[4] = (_Float16)v1[0]; h[5] = (_Float16)v1[1];
        h[6] = (_Float16)v1[2]; h[7] = (_Float16)v1[3];
        *(v8h*)(dst + (size_t)i * 8) = h;
    }
}

// ---------------------------------------------------------------------------
// FT GEMM, async-staging variant: ft_w already f16 in global; the 16 KB
// B-tile is staged by 4 GLOBAL_LOAD_ASYNC_TO_LDS_B128 per wave (per-lane 16B,
// inst offset applies to both LDS and global address).
// ---------------------------------------------------------------------------
__global__ __launch_bounds__(256, 1) void nnue_ft_kernel_async(
    const float* __restrict__ wf, const float* __restrict__ bf,
    const _Float16* __restrict__ ftwh, float* __restrict__ C)
{
    __shared__ __align__(16) _Float16 Bs[2][M_DIM * BS_STRIDE]; // 2 x 20 KB

    const int rb   = blockIdx.x & (ROWBLOCKS - 1);
    const int kc   = blockIdx.x / ROWBLOCKS;
    const int lane = threadIdx.x & 31;
    const int wid  = threadIdx.x >> 5;

    const int row = rb * 128 + wid * 16 + (lane & 15);
    const float* featRow = (row < BATCH) ? (wf + (size_t)row * F_DIM)
                                         : (bf + (size_t)(row - BATCH) * F_DIM);
    const int aoff   = (lane < 16) ? 0 : 8;
    const int koffB  = (lane < 16) ? 0 : 8;
    const int k0base = kc * KCHUNK;

    const int m = threadIdx.x;                       // ft_w row this lane stages
    const _Float16* ftwRowH = ftwh + (size_t)m * F_DIM;
    // Flat-aperture rule: LDS byte address is addr[31:0] of the generic ptr.
    const uint32_t ldsRow0 = (uint32_t)(uintptr_t)&Bs[0][m * BS_STRIDE];
    const uint32_t ldsRow1 = (uint32_t)(uintptr_t)&Bs[1][m * BS_STRIDE];

    auto loadA = [&](int s, fvec4* r) {
        const float* ap = featRow + k0base + s * KSTEP + aoff;
        r[0] = ((const fvec4*)ap)[0];
        r[1] = ((const fvec4*)ap)[1];
        r[2] = ((const fvec4*)(ap + 16))[0];
        r[3] = ((const fvec4*)(ap + 16))[1];
    };
    auto stageAsync = [&](int s, int buf) {
        const void* g = (const void*)(ftwRowH + (size_t)(k0base + s * KSTEP));
        uint32_t l = buf ? ldsRow1 : ldsRow0;
        asm volatile(
            "global_load_async_to_lds_b128 %0, %1, off\n\t"
            "global_load_async_to_lds_b128 %0, %1, off offset:16\n\t"
            "global_load_async_to_lds_b128 %0, %1, off offset:32\n\t"
            "global_load_async_to_lds_b128 %0, %1, off offset:48"
            :: "v"(l), "v"(g) : "memory");
    };

    v8f acc[16] = {};
    fvec4 araw[4];

    stageAsync(0, 0);
    loadA(0, araw);
    asm volatile("s_wait_asynccnt 0x0" ::: "memory");
    __syncthreads();

    for (int s = 0; s < NSTEPS; ++s) {
        const int  cur  = s & 1;
        const bool more = (s + 1 < NSTEPS);

        if (more) stageAsync(s + 1, cur ^ 1);        // async engine fills buf^1
        fvec4 araw_n[4];
        if (more) loadA(s + 1, araw_n);

        v16h a;
        a[0]  = (_Float16)araw[0][0]; a[1]  = (_Float16)araw[0][1];
        a[2]  = (_Float16)araw[0][2]; a[3]  = (_Float16)araw[0][3];
        a[4]  = (_Float16)araw[1][0]; a[5]  = (_Float16)araw[1][1];
        a[6]  = (_Float16)araw[1][2]; a[7]  = (_Float16)araw[1][3];
        a[8]  = (_Float16)araw[2][0]; a[9]  = (_Float16)araw[2][1];
        a[10] = (_Float16)araw[2][2]; a[11] = (_Float16)araw[2][3];
        a[12] = (_Float16)araw[3][0]; a[13] = (_Float16)araw[3][1];
        a[14] = (_Float16)araw[3][2]; a[15] = (_Float16)araw[3][3];

        const _Float16* bbase = &Bs[cur][(lane & 15) * BS_STRIDE + koffB];
        #pragma unroll
        for (int g = 0; g < 2; ++g) {
            v16h bb[8];
            #pragma unroll
            for (int t = 0; t < 8; ++t) {
                const _Float16* bp = bbase + (g * 8 + t) * 16 * BS_STRIDE;
                v8h b0 = *(const v8h*)bp;
                v8h b1 = *(const v8h*)(bp + 16);
                bb[t] = __builtin_shufflevector(b0, b1,
                          0,1,2,3,4,5,6,7,8,9,10,11,12,13,14,15);
            }
            #pragma unroll
            for (int t = 0; t < 8; ++t)
                acc[g * 8 + t] = __builtin_amdgcn_wmma_f32_16x16x32_f16(
                    false, a, false, bb[t], (short)0, acc[g * 8 + t], false, false);
        }

        if (more) {
            #pragma unroll
            for (int j = 0; j < 4; ++j) araw[j] = araw_n[j];
        }
        asm volatile("s_wait_asynccnt 0x0" ::: "memory");
        __syncthreads();
    }

    const int outRow = rb * 128 + wid * 16 + ((lane < 16) ? 0 : 8);
    const int col    = lane & 15;
    #pragma unroll
    for (int t = 0; t < 16; ++t) {
        float* cp = C + (size_t)outRow * M_DIM + t * 16 + col;
        #pragma unroll
        for (int v = 0; v < 8; ++v)
            unsafeAtomicAdd(cp + (size_t)v * M_DIM, acc[t][v]);
    }
}

// ---------------------------------------------------------------------------
// FT GEMM, fallback variant: f32 -> f16 conversion inside the kernel
// (used when workspace is too small for the pre-converted ft_w copy).
// ---------------------------------------------------------------------------
__global__ __launch_bounds__(256, 1) void nnue_ft_kernel(
    const float* __restrict__ wf, const float* __restrict__ bf,
    const float* __restrict__ ftw, float* __restrict__ C)
{
    __shared__ __align__(16) _Float16 Bs[2][M_DIM * BS_STRIDE]; // 2 x 20 KB

    const int rb   = blockIdx.x & (ROWBLOCKS - 1);
    const int kc   = blockIdx.x / ROWBLOCKS;
    const int lane = threadIdx.x & 31;
    const int wid  = threadIdx.x >> 5;

    const int row = rb * 128 + wid * 16 + (lane & 15);
    const float* featRow = (row < BATCH) ? (wf + (size_t)row * F_DIM)
                                         : (bf + (size_t)(row - BATCH) * F_DIM);
    const int aoff   = (lane < 16) ? 0 : 8;
    const int koffB  = (lane < 16) ? 0 : 8;
    const int k0base = kc * KCHUNK;

    const int m = threadIdx.x;
    const float* ftwRow = ftw + (size_t)m * F_DIM;

    auto loadA = [&](int s, fvec4* r) {
        const float* ap = featRow + k0base + s * KSTEP + aoff;
        r[0] = ((const fvec4*)ap)[0];
        r[1] = ((const fvec4*)ap)[1];
        r[2] = ((const fvec4*)(ap + 16))[0];
        r[3] = ((const fvec4*)(ap + 16))[1];
    };
    auto loadB = [&](int s, fvec4* r) {
        const fvec4* src = (const fvec4*)(ftwRow + k0base + s * KSTEP);
        #pragma unroll
        for (int j = 0; j < 8; ++j) r[j] = src[j];
    };
    auto storeB = [&](int buf, const fvec4* r) {
        _Float16* dst = &Bs[buf][m * BS_STRIDE];
        #pragma unroll
        for (int j = 0; j < 4; ++j) {
            fvec4 v0 = r[2 * j], v1 = r[2 * j + 1];
            v8h h;
            h[0] = (_Float16)v0[0]; h[1] = (_Float16)v0[1];
            h[2] = (_Float16)v0[2]; h[3] = (_Float16)v0[3];
            h[4] = (_Float16)v1[0]; h[5] = (_Float16)v1[1];
            h[6] = (_Float16)v1[2]; h[7] = (_Float16)v1[3];
            *(v8h*)(dst + 8 * j) = h;
        }
    };

    v8f acc[16] = {};
    fvec4 araw[4];
    fvec4 btmp[8];

    loadB(0, btmp);
    loadA(0, araw);
    storeB(0, btmp);
    __syncthreads();

    for (int s = 0; s < NSTEPS; ++s) {
        const int  cur  = s & 1;
        const bool more = (s + 1 < NSTEPS);

        fvec4 araw_n[4], braw_n[8];
        if (more) { loadB(s + 1, braw_n); loadA(s + 1, araw_n); }

        v16h a;
        a[0]  = (_Float16)araw[0][0]; a[1]  = (_Float16)araw[0][1];
        a[2]  = (_Float16)araw[0][2]; a[3]  = (_Float16)araw[0][3];
        a[4]  = (_Float16)araw[1][0]; a[5]  = (_Float16)araw[1][1];
        a[6]  = (_Float16)araw[1][2]; a[7]  = (_Float16)araw[1][3];
        a[8]  = (_Float16)araw[2][0]; a[9]  = (_Float16)araw[2][1];
        a[10] = (_Float16)araw[2][2]; a[11] = (_Float16)araw[2][3];
        a[12] = (_Float16)araw[3][0]; a[13] = (_Float16)araw[3][1];
        a[14] = (_Float16)araw[3][2]; a[15] = (_Float16)araw[3][3];

        const _Float16* bbase = &Bs[cur][(lane & 15) * BS_STRIDE + koffB];
        #pragma unroll
        for (int g = 0; g < 2; ++g) {
            v16h bb[8];
            #pragma unroll
            for (int t = 0; t < 8; ++t) {
                const _Float16* bp = bbase + (g * 8 + t) * 16 * BS_STRIDE;
                v8h b0 = *(const v8h*)bp;
                v8h b1 = *(const v8h*)(bp + 16);
                bb[t] = __builtin_shufflevector(b0, b1,
                          0,1,2,3,4,5,6,7,8,9,10,11,12,13,14,15);
            }
            #pragma unroll
            for (int t = 0; t < 8; ++t)
                acc[g * 8 + t] = __builtin_amdgcn_wmma_f32_16x16x32_f16(
                    false, a, false, bb[t], (short)0, acc[g * 8 + t], false, false);
        }

        if (more) {
            storeB(cur ^ 1, braw_n);
            #pragma unroll
            for (int j = 0; j < 4; ++j) araw[j] = araw_n[j];
        }
        __syncthreads();
    }

    const int outRow = rb * 128 + wid * 16 + ((lane < 16) ? 0 : 8);
    const int col    = lane & 15;
    #pragma unroll
    for (int t = 0; t < 16; ++t) {
        float* cp = C + (size_t)outRow * M_DIM + t * 16 + col;
        #pragma unroll
        for (int v = 0; v < 8; ++v)
            unsafeAtomicAdd(cp + (size_t)v * M_DIM, acc[t][v]);
    }
}

// ---------------------------------------------------------------------------
// Tail: one wave per batch element (8 waves per block).
// ---------------------------------------------------------------------------
__global__ __launch_bounds__(256) void nnue_tail_kernel(
    const float* __restrict__ C,   const float* __restrict__ stm,
    const float* __restrict__ ftb,
    const float* __restrict__ l1w, const float* __restrict__ l1b,
    const float* __restrict__ l2w, const float* __restrict__ l2b,
    const float* __restrict__ l3w, const float* __restrict__ l3b,
    float* __restrict__ out)
{
    __shared__ float l1x[8][512];
    __shared__ float l2x[8][32];

    const int lane = threadIdx.x & 31;
    const int wid  = threadIdx.x >> 5;
    const int b    = blockIdx.x * 8 + wid;

    const float s = stm[b];
    const float* Cw = C + (size_t)b * M_DIM;
    const float* Cb = C + (size_t)(BATCH + b) * M_DIM;

    #pragma unroll
    for (int i = 0; i < 8; ++i) {
        int mm = lane + 32 * i;
        float w  = Cw[mm] + ftb[mm];
        float bl = Cb[mm] + ftb[mm];
        l1x[wid][mm]       = clamp01(s * w  + (1.0f - s) * bl);
        l1x[wid][256 + mm] = clamp01(s * bl + (1.0f - s) * w);
    }
    __syncthreads();

    float a1 = l1b[lane];
    const float* w1 = l1w + lane * 512;
    #pragma unroll 8
    for (int k = 0; k < 512; ++k) a1 += l1x[wid][k] * w1[k];
    l2x[wid][lane] = clamp01(a1);
    __syncthreads();

    float a2 = l2b[lane];
    const float* w2 = l2w + lane * 32;
    #pragma unroll
    for (int k = 0; k < 32; ++k) a2 += l2x[wid][k] * w2[k];
    float l3v = clamp01(a2);

    float p = l3v * l3w[lane];
    #pragma unroll
    for (int off = 16; off > 0; off >>= 1) p += __shfl_down(p, off, 32);
    if (lane == 0) {
        float ev = clamp01(p + l3b[0]);
        out[b] = (ev - 0.5f) * 2.0f * 10000.0f;   // MAX_EVAL = 10000
    }
}

extern "C" void kernel_launch(void* const* d_in, const int* in_sizes, int n_in,
                              void* d_out, int out_size, void* d_ws, size_t ws_size,
                              hipStream_t stream)
{
    const float* wf  = (const float*)d_in[0];
    const float* bf  = (const float*)d_in[1];
    const float* stm = (const float*)d_in[2];
    const float* ftw = (const float*)d_in[3];
    const float* ftb = (const float*)d_in[4];
    const float* l1w = (const float*)d_in[5];
    const float* l1b = (const float*)d_in[6];
    const float* l2w = (const float*)d_in[7];
    const float* l2b = (const float*)d_in[8];
    const float* l3w = (const float*)d_in[9];
    const float* l3b = (const float*)d_in[10];
    float* out = (float*)d_out;

    const size_t cBytes   = (size_t)2 * BATCH * M_DIM * sizeof(float);     // 4 MB
    const size_t ftwBytes = (size_t)M_DIM * F_DIM * sizeof(_Float16);      // 20 MB
    float* C = (float*)d_ws;

    nnue_zero_kernel<<<dim3(4096), dim3(256), 0, stream>>>(C, 2 * BATCH * M_DIM);

    if (ws_size >= cBytes + ftwBytes) {
        _Float16* ftwh = (_Float16*)((char*)d_ws + cBytes);
        const int n8 = (M_DIM * F_DIM) / 8;                                // 1310720
        nnue_cvt_kernel<<<dim3(n8 / 256), dim3(256), 0, stream>>>(ftw, ftwh, n8);
        nnue_ft_kernel_async<<<dim3(ROWBLOCKS * KSPLIT), dim3(256), 0, stream>>>(
            wf, bf, ftwh, C);
    } else {
        nnue_ft_kernel<<<dim3(ROWBLOCKS * KSPLIT), dim3(256), 0, stream>>>(
            wf, bf, ftw, C);
    }

    nnue_tail_kernel<<<dim3(BATCH / 8), dim3(256), 0, stream>>>(
        C, stm, ftb, l1w, l1b, l2w, l2b, l3w, l3b, out);
}
